// CTMCore_35184372089261
// MI455X (gfx1250) — matrix operand order repaired and checked by reading, not compiled, not run on hardware
//
#include <hip/hip_runtime.h>

// ---------------- dims ----------------
#define Bz 64
#define Nz 1024
#define Ez 512
#define Dz 1024
#define Mz 16
#define Tz 12
#define Pz 2048
#define HSz 2048
#define HNz 64
#define Cz 4
#define NHz 8
#define DHz 64

typedef unsigned short u16;
typedef __attribute__((ext_vector_type(4)))  float v4f;
typedef __attribute__((ext_vector_type(8)))  float v8f;
typedef __attribute__((ext_vector_type(8)))  u16   v8us;
typedef __attribute__((ext_vector_type(16))) u16   v16us;
typedef __attribute__((ext_vector_type(16))) __bf16 v16bf;

union BFrag { v16us u; v16bf b; };

__device__ __forceinline__ float bf2f(u16 u) {
    union { unsigned i; float f; } c; c.i = ((unsigned)u) << 16; return c.f;
}
__device__ __forceinline__ u16 f2bf(float f) {
    union { float f; unsigned i; } c; c.f = f;
    unsigned r = c.i + 0x7FFFu + ((c.i >> 16) & 1u);
    return (u16)(r >> 16);
}
__device__ __forceinline__ float sigf(float x) { return 1.f / (1.f + __expf(-x)); }

// ---------------- generic WMMA bf16 GEMM: C[M,N] = A[M,K] @ W[N,K]^T + bias ----------------
// grid = (N/16, M/64), block = 128 (4 waves, one 16x16 tile each)
__global__ __launch_bounds__(128)
void gemm_bf16_kernel(const u16* __restrict__ A, const u16* __restrict__ W,
                      const float* __restrict__ bias,
                      float* __restrict__ Cf, u16* __restrict__ Cbf,
                      int Ndim, int Kdim)
{
    int lane = threadIdx.x & 31;
    int wave = threadIdx.x >> 5;
    int tn = blockIdx.x;
    int tm = blockIdx.y * 4 + wave;
    int rc = lane & 15;
    int half = lane >> 4;

    const u16* ap = A + (size_t)(tm * 16 + rc) * Kdim + half * 8;
    const u16* bp = W + (size_t)(tn * 16 + rc) * Kdim + half * 8;

    v8f acc = {};
    for (int kb = 0; kb < Kdim; kb += 32) {
        v8us a0 = *(const v8us*)(ap + kb);
        v8us a1 = *(const v8us*)(ap + kb + 16);
        v8us b0 = *(const v8us*)(bp + kb);
        v8us b1 = *(const v8us*)(bp + kb + 16);
        BFrag af, bf;
        af.u = __builtin_shufflevector(a0, a1, 0,1,2,3,4,5,6,7,8,9,10,11,12,13,14,15);
        bf.u = __builtin_shufflevector(b0, b1, 0,1,2,3,4,5,6,7,8,9,10,11,12,13,14,15);
        acc = __builtin_amdgcn_wmma_f32_16x16x32_bf16(false, af.b, false, bf.b,
                                                      (short)0, acc, false, false);
    }
    int col = tn * 16 + rc;
    float bv = bias ? bias[col] : 0.f;
#pragma unroll
    for (int r = 0; r < 8; ++r) {
        int row = tm * 16 + r + half * 8;
        float v = acc[r] + bv;
        if (Cf)  Cf[(size_t)row * Ndim + col] = v;
        if (Cbf) Cbf[(size_t)row * Ndim + col] = f2bf(v);
    }
}

// ---------------- K/V projection: Out[b,h,n,dh] = bf16(X[b*N+n,:] @ W[col,:]^T + bias) ----------------
// X fp32 [B*N, E]; grid = (E/16, (B*N)/64), block = 128
__global__ __launch_bounds__(128)
void kv_gemm_kernel(const float* __restrict__ X, const u16* __restrict__ W,
                    const float* __restrict__ bias, u16* __restrict__ Out, int Kdim)
{
    int lane = threadIdx.x & 31;
    int wave = threadIdx.x >> 5;
    int tn = blockIdx.x;
    int tm = blockIdx.y * 4 + wave;
    int rc = lane & 15;
    int half = lane >> 4;

    const float* ap = X + (size_t)(tm * 16 + rc) * Kdim + half * 8;
    const u16*   bp = W + (size_t)(tn * 16 + rc) * Kdim + half * 8;

    v8f acc = {};
    for (int kb = 0; kb < Kdim; kb += 32) {
        v4f f0 = *(const v4f*)(ap + kb);
        v4f f1 = *(const v4f*)(ap + kb + 4);
        v4f f2 = *(const v4f*)(ap + kb + 16);
        v4f f3 = *(const v4f*)(ap + kb + 20);
        BFrag af, bf;
#pragma unroll
        for (int j = 0; j < 4; ++j) {
            af.u[j]      = f2bf(f0[j]);
            af.u[4 + j]  = f2bf(f1[j]);
            af.u[8 + j]  = f2bf(f2[j]);
            af.u[12 + j] = f2bf(f3[j]);
        }
        v8us b0 = *(const v8us*)(bp + kb);
        v8us b1 = *(const v8us*)(bp + kb + 16);
        bf.u = __builtin_shufflevector(b0, b1, 0,1,2,3,4,5,6,7,8,9,10,11,12,13,14,15);
        acc = __builtin_amdgcn_wmma_f32_16x16x32_bf16(false, af.b, false, bf.b,
                                                      (short)0, acc, false, false);
    }
    int col = tn * 16 + rc;
    float bv = bias[col];
    int h = col >> 6, dd = col & 63;
#pragma unroll
    for (int r = 0; r < 8; ++r) {
        int row = tm * 16 + r + half * 8;
        int b = row >> 10, n = row & 1023;
        Out[(((size_t)b * NHz + h) * Nz + n) * DHz + dd] = f2bf(acc[r] + bv);
    }
}

// ---------------- attention per (b,h): softmax(q.k/sqrt(dh)) @ v ----------------
__global__ __launch_bounds__(256)
void attn_kernel(const float* __restrict__ qh, const u16* __restrict__ K,
                 const u16* __restrict__ V, u16* __restrict__ Obf)
{
    __shared__ float sc[Nz];
    __shared__ float red[256];
    __shared__ float qs[DHz];
    int bh = blockIdx.x;
    int b = bh >> 3, h = bh & 7;
    int tid = threadIdx.x;
    if (tid < DHz) qs[tid] = qh[(size_t)b * Ez + h * DHz + tid];
    __syncthreads();

    const u16* kb_ = K + (size_t)bh * Nz * DHz;
    float myscore[4];
    float lmax = -1e30f;
#pragma unroll
    for (int i = 0; i < 4; ++i) {
        int n = tid + i * 256;
        const u16* kr = kb_ + (size_t)n * DHz;
        float s = 0.f;
        for (int d = 0; d < DHz; d += 8) {
            v8us kk = *(const v8us*)(kr + d);
#pragma unroll
            for (int j = 0; j < 8; ++j) s += qs[d + j] * bf2f(kk[j]);
        }
        s *= 0.125f;  // 1/sqrt(64)
        myscore[i] = s;
        lmax = fmaxf(lmax, s);
    }
    red[tid] = lmax; __syncthreads();
    for (int k = 128; k > 0; k >>= 1) {
        if (tid < k) red[tid] = fmaxf(red[tid], red[tid + k]);
        __syncthreads();
    }
    float gmax = red[0]; __syncthreads();
    float lsum = 0.f;
#pragma unroll
    for (int i = 0; i < 4; ++i) {
        int n = tid + i * 256;
        float e = __expf(myscore[i] - gmax);
        sc[n] = e; lsum += e;
    }
    red[tid] = lsum; __syncthreads();
    for (int k = 128; k > 0; k >>= 1) {
        if (tid < k) red[tid] += red[tid + k];
        __syncthreads();
    }
    float inv = 1.f / red[0]; __syncthreads();

    int d = tid & 63, chunk = tid >> 6;
    const u16* vb_ = V + (size_t)bh * Nz * DHz;
    float acc = 0.f;
    for (int n = chunk * 256; n < chunk * 256 + 256; ++n)
        acc += sc[n] * bf2f(vb_[(size_t)n * DHz + d]);
    red[tid] = acc; __syncthreads();
    if (tid < 64) {
        float o = (red[tid] + red[tid + 64] + red[tid + 128] + red[tid + 192]) * inv;
        Obf[(size_t)b * Ez + h * DHz + d] = f2bf(o);
    }
}

// ---------------- LayerNorm; optionally scatter bf16 into concat buffers ----------------
__global__ __launch_bounds__(256)
void ln_kernel(const float* __restrict__ X, const float* __restrict__ g,
               const float* __restrict__ bta, float* __restrict__ Y,
               u16* __restrict__ U1, u16* __restrict__ U2,
               int cols, int ustride, int uoff)
{
    __shared__ float red[256];
    int row = blockIdx.x, tid = threadIdx.x;
    const float* x = X + (size_t)row * cols;
    float s = 0.f;
    for (int c = tid; c < cols; c += 256) s += x[c];
    red[tid] = s; __syncthreads();
    for (int k = 128; k > 0; k >>= 1) { if (tid < k) red[tid] += red[tid + k]; __syncthreads(); }
    float mean = red[0] / (float)cols; __syncthreads();
    float v = 0.f;
    for (int c = tid; c < cols; c += 256) { float d = x[c] - mean; v += d * d; }
    red[tid] = v; __syncthreads();
    for (int k = 128; k > 0; k >>= 1) { if (tid < k) red[tid] += red[tid + k]; __syncthreads(); }
    float inv = rsqrtf(red[0] / (float)cols + 1e-5f);
    for (int c = tid; c < cols; c += 256) {
        float y = (x[c] - mean) * inv * g[c] + bta[c];
        if (Y)  Y[(size_t)row * cols + c] = y;
        u16 bb = f2bf(y);
        if (U1) U1[(size_t)row * ustride + uoff + c] = bb;
        if (U2) U2[(size_t)row * ustride + uoff + c] = bb;
    }
}

// ---------------- gated fusion: z_bar = (1-g)*zn + g*tanh(h) -> fused_bf[:, :D] ----------------
__global__ void fuse_kernel(const float* __restrict__ gpre, const float* __restrict__ hpre,
                            const float* __restrict__ zn, u16* __restrict__ fused)
{
    int i = blockIdx.x * 256 + threadIdx.x;   // 0 .. B*D-1
    int b = i >> 10, c = i & 1023;
    float gg = sigf(gpre[i]);
    float zb = (1.f - gg) * zn[i] + gg * tanhf(hpre[i]);
    fused[(size_t)b * (Dz + Ez) + c] = f2bf(zb);
}

// ---------------- exact GELU -> bf16 ----------------
__global__ void gelu_kernel(const float* __restrict__ x, u16* __restrict__ y)
{
    int i = blockIdx.x * 256 + threadIdx.x;   // B*HS
    float v = x[i];
    float gv = 0.5f * v * (1.f + erff(v * 0.70710678f));
    y[i] = f2bf(gv);
}

// ---------------- NLM: per-neuron SuperLinear + GLU twice ----------------
__global__ __launch_bounds__(256)
void nlm_kernel(const float* __restrict__ preh, const float* __restrict__ w1,
                const float* __restrict__ b1, const float* __restrict__ w2,
                const float* __restrict__ b2, float* __restrict__ z,
                float* __restrict__ zt_out, int t)
{
    int i = blockIdx.x * 256 + threadIdx.x;   // b*D + d
    int b = i >> 10, d = i & 1023;
    int cnt = t + 1;
    float pv[Tz];
    for (int j = 0; j < cnt; ++j) pv[j] = preh[(size_t)j * (Bz * Dz) + i];
    const float* W1 = w1 + (size_t)d * Mz * (2 * HNz);
    const float* B1 = b1 + (size_t)d * (2 * HNz);
    const float* W2 = w2 + (size_t)d * HNz * 2;
    float z0 = 0.f, z1 = 0.f;
    for (int hh = 0; hh < HNz; ++hh) {
        float a = B1[hh], bb = B1[HNz + hh];
        for (int j = 0; j < cnt; ++j) {
            int m = Mz - 1 - t + j;           // history slot in shifted A
            float p = pv[j];
            a  += p * W1[m * (2 * HNz) + hh];
            bb += p * W1[m * (2 * HNz) + HNz + hh];
        }
        float h1 = a * sigf(bb);
        z0 += h1 * W2[hh * 2 + 0];
        z1 += h1 * W2[hh * 2 + 1];
    }
    z0 += b2[d * 2 + 0];
    z1 += b2[d * 2 + 1];
    float zv = z0 * sigf(z1);
    z[i] = zv;
    zt_out[(size_t)b * (Tz * Dz) + (size_t)t * Dz + d] = zv;
}

// ---------------- decayed pair synchronisation ----------------
__global__ void sync_kernel(const float* __restrict__ z, const float* __restrict__ decays,
                            const int* __restrict__ al, const int* __restrict__ ar,
                            const int* __restrict__ ol, const int* __restrict__ orr,
                            float* __restrict__ a_a, float* __restrict__ b_a,
                            float* __restrict__ a_o, float* __restrict__ b_o,
                            u16* __restrict__ s_a_bf, float* __restrict__ s_o)
{
    int i = blockIdx.x * 256 + threadIdx.x;   // b*P + p
    int b = i >> 11, p = i & 2047;
    const float* zb = z + (size_t)b * Dz;
    float da = decays[p];
    float na  = da * a_a[i] + zb[al[p]] * zb[ar[p]];
    float nba = da * b_a[i] + 1.f;
    a_a[i] = na; b_a[i] = nba;
    s_a_bf[i] = f2bf(na * rsqrtf(nba + 1e-8f));
    float dO = decays[Pz + p];
    float no  = dO * a_o[i] + zb[ol[p]] * zb[orr[p]];
    float nbo = dO * b_o[i] + 1.f;
    a_o[i] = no; b_o[i] = nbo;
    s_o[i] = no * rsqrtf(nbo + 1e-8f);
}

// ---------------- head + certainty ----------------
__global__ __launch_bounds__(256)
void head_kernel(const float* __restrict__ s_o, const float* __restrict__ hw,
                 const float* __restrict__ hb, float* __restrict__ out, int t)
{
    __shared__ float red[Cz][256];
    int b = blockIdx.x, tid = threadIdx.x;
    const float* s = s_o + (size_t)b * Pz;
    float acc[Cz] = {0.f, 0.f, 0.f, 0.f};
    for (int p = tid; p < Pz; p += 256) {
        float sv = s[p];
#pragma unroll
        for (int c = 0; c < Cz; ++c) acc[c] += sv * hw[c * Pz + p];
    }
#pragma unroll
    for (int c = 0; c < Cz; ++c) red[c][tid] = acc[c];
    __syncthreads();
    for (int k = 128; k > 0; k >>= 1) {
        if (tid < k) {
#pragma unroll
            for (int c = 0; c < Cz; ++c) red[c][tid] += red[c][tid + k];
        }
        __syncthreads();
    }
    if (tid == 0) {
        float lg[Cz];
#pragma unroll
        for (int c = 0; c < Cz; ++c) lg[c] = red[c][0] + hb[c];
        float mx = fmaxf(fmaxf(lg[0], lg[1]), fmaxf(lg[2], lg[3]));
        float e[Cz], sum = 0.f;
#pragma unroll
        for (int c = 0; c < Cz; ++c) { e[c] = __expf(lg[c] - mx); sum += e[c]; }
        float ent = 0.f;
#pragma unroll
        for (int c = 0; c < Cz; ++c) {
            float pr = fmaxf(e[c] / sum, 1e-8f);
            ent -= pr * __logf(pr);
        }
#pragma unroll
        for (int c = 0; c < Cz; ++c) out[b * (Cz * Tz) + c * Tz + t] = lg[c];
        out[Bz * Cz * Tz + b * Tz + t] = 1.f - ent * (1.f / 1.3862943611f);
    }
}

// ---------------- helpers: convert / decay / init ----------------
__global__ void convert_kernel(const float* __restrict__ src, u16* __restrict__ dst, int n)
{
    int i = blockIdx.x * 256 + threadIdx.x;
    if (i < n) dst[i] = f2bf(src[i]);
}
__global__ void decay_kernel(const float* __restrict__ ra, const float* __restrict__ ro,
                             float* __restrict__ decays)
{
    int p = blockIdx.x * 256 + threadIdx.x;
    if (p >= Pz) return;
    float xa = ra[p], xo = ro[p];
    float spa = xa > 0.f ? xa + log1pf(__expf(-xa)) : log1pf(__expf(xa));
    float spo = xo > 0.f ? xo + log1pf(__expf(-xo)) : log1pf(__expf(xo));
    decays[p] = __expf(-spa);
    decays[Pz + p] = __expf(-spo);
}
__global__ void init_kernel(float* __restrict__ z, float* __restrict__ a_a, float* __restrict__ b_a,
                            float* __restrict__ a_o, float* __restrict__ b_o, u16* __restrict__ s_a_bf)
{
    int i = blockIdx.x * 256 + threadIdx.x;
    if (i < Bz * Pz) { a_a[i] = 0.f; b_a[i] = 1.f; a_o[i] = 0.f; b_o[i] = 1.f; s_a_bf[i] = 0; }
    if (i < Bz * Dz) z[i] = 0.f;
}

extern "C" void kernel_launch(void* const* d_in, const int* in_sizes, int n_in,
                              void* d_out, int out_size, void* d_ws, size_t ws_size,
                              hipStream_t stream)
{
    (void)in_sizes; (void)n_in; (void)out_size; (void)ws_size;

    const float* kv        = (const float*)d_in[0];
    const float* raw_r_act = (const float*)d_in[1];
    const float* raw_r_out = (const float*)d_in[2];
    const float* q_w       = (const float*)d_in[3];
    const float* q_b       = (const float*)d_in[4];
    const float* in_proj_w = (const float*)d_in[5];
    const float* in_proj_b = (const float*)d_in[6];
    const float* out_proj_w= (const float*)d_in[7];
    const float* out_proj_b= (const float*)d_in[8];
    const float* z_ln_g    = (const float*)d_in[9];
    const float* z_ln_b    = (const float*)d_in[10];
    const float* o_ln_g    = (const float*)d_in[11];
    const float* o_ln_b    = (const float*)d_in[12];
    const float* gate_w    = (const float*)d_in[13];
    const float* gate_b    = (const float*)d_in[14];
    const float* cand_w    = (const float*)d_in[15];
    const float* cand_b    = (const float*)d_in[16];
    const float* syn1_w    = (const float*)d_in[17];
    const float* syn1_b    = (const float*)d_in[18];
    const float* syn2_w    = (const float*)d_in[19];
    const float* syn2_b    = (const float*)d_in[20];
    const float* nlm1_w    = (const float*)d_in[21];
    const float* nlm1_b    = (const float*)d_in[22];
    const float* nlm2_w    = (const float*)d_in[23];
    const float* nlm2_b    = (const float*)d_in[24];
    const float* head_w    = (const float*)d_in[25];
    const float* head_b    = (const float*)d_in[26];
    const int*   act_left  = (const int*)d_in[27];
    const int*   act_right = (const int*)d_in[28];
    const int*   out_left  = (const int*)d_in[29];
    const int*   out_right = (const int*)d_in[30];
    float* out = (float*)d_out;

    char* ws = (char*)d_ws;
    size_t off = 0;
    auto alloc = [&](size_t bytes) -> char* {
        char* p = ws + off;
        off = (off + bytes + 255) & ~(size_t)255;
        return p;
    };

    u16* k_bf    = (u16*)alloc((size_t)Bz * NHz * Nz * DHz * 2);
    u16* v_bf    = (u16*)alloc((size_t)Bz * NHz * Nz * DHz * 2);
    u16* qw_bf   = (u16*)alloc((size_t)Ez * Pz * 2);
    u16* wq_bf   = (u16*)alloc((size_t)Ez * Ez * 2);
    u16* wk_bf   = (u16*)alloc((size_t)Ez * Ez * 2);
    u16* wv_bf   = (u16*)alloc((size_t)Ez * Ez * 2);
    u16* op_bf   = (u16*)alloc((size_t)Ez * Ez * 2);
    u16* gate_bf = (u16*)alloc((size_t)Dz * (Dz + Ez) * 2);
    u16* cand_bf = (u16*)alloc((size_t)Dz * (Dz + Ez) * 2);
    u16* syn1_bf = (u16*)alloc((size_t)HSz * (Dz + Ez) * 2);
    u16* syn2_bf = (u16*)alloc((size_t)Dz * HSz * 2);
    float* decays = (float*)alloc(2 * Pz * 4);
    float* z      = (float*)alloc((size_t)Bz * Dz * 4);
    float* a_a    = (float*)alloc((size_t)Bz * Pz * 4);
    float* b_a    = (float*)alloc((size_t)Bz * Pz * 4);
    float* a_o    = (float*)alloc((size_t)Bz * Pz * 4);
    float* b_o    = (float*)alloc((size_t)Bz * Pz * 4);
    float* s_o    = (float*)alloc((size_t)Bz * Pz * 4);
    u16* s_a_bf   = (u16*)alloc((size_t)Bz * Pz * 2);
    float* qf     = (float*)alloc((size_t)Bz * Ez * 4);
    u16* q_bf16   = (u16*)alloc((size_t)Bz * Ez * 2);
    float* qhp    = (float*)alloc((size_t)Bz * Ez * 4);
    u16* o_bf     = (u16*)alloc((size_t)Bz * Ez * 2);
    float* o2     = (float*)alloc((size_t)Bz * Ez * 4);
    float* zn     = (float*)alloc((size_t)Bz * Dz * 4);
    u16* u_bf     = (u16*)alloc((size_t)Bz * (Dz + Ez) * 2);
    u16* fused_bf = (u16*)alloc((size_t)Bz * (Dz + Ez) * 2);
    float* gpre   = (float*)alloc((size_t)Bz * Dz * 4);
    float* hpre   = (float*)alloc((size_t)Bz * Dz * 4);
    float* s1     = (float*)alloc((size_t)Bz * HSz * 4);
    u16* s1g      = (u16*)alloc((size_t)Bz * HSz * 2);
    float* preh   = (float*)alloc((size_t)Tz * Bz * Dz * 4);

    auto conv = [&](const float* s, u16* d, int n) {
        convert_kernel<<<(n + 255) / 256, 256, 0, stream>>>(s, d, n);
    };
    // one-time weight/bias precompute
    conv(q_w, qw_bf, Ez * Pz);
    conv(in_proj_w,               wq_bf, Ez * Ez);
    conv(in_proj_w + Ez * Ez,     wk_bf, Ez * Ez);
    conv(in_proj_w + 2 * Ez * Ez, wv_bf, Ez * Ez);
    conv(out_proj_w, op_bf, Ez * Ez);
    conv(gate_w, gate_bf, Dz * (Dz + Ez));
    conv(cand_w, cand_bf, Dz * (Dz + Ez));
    conv(syn1_w, syn1_bf, HSz * (Dz + Ez));
    conv(syn2_w, syn2_bf, Dz * HSz);
    decay_kernel<<<Pz / 256, 256, 0, stream>>>(raw_r_act, raw_r_out, decays);
    init_kernel<<<(Bz * Pz) / 256, 256, 0, stream>>>(z, a_a, b_a, a_o, b_o, s_a_bf);

    // tick-invariant K/V projection (WMMA), bf16 output in [B,NH,N,dh]
    dim3 kvg(Ez / 16, (Bz * Nz) / 64);
    kv_gemm_kernel<<<kvg, 128, 0, stream>>>(kv, wk_bf, in_proj_b + Ez,     k_bf, Ez);
    kv_gemm_kernel<<<kvg, 128, 0, stream>>>(kv, wv_bf, in_proj_b + 2 * Ez, v_bf, Ez);

    for (int t = 0; t < Tz; ++t) {
        // q = s_a @ q_w.T + q_b ; qh = q @ Wq.T + bq
        gemm_bf16_kernel<<<dim3(Ez / 16, 1), 128, 0, stream>>>(s_a_bf, qw_bf, q_b, qf, q_bf16, Ez, Pz);
        gemm_bf16_kernel<<<dim3(Ez / 16, 1), 128, 0, stream>>>(q_bf16, wq_bf, in_proj_b, qhp, nullptr, Ez, Ez);
        // attention
        attn_kernel<<<Bz * NHz, 256, 0, stream>>>(qhp, k_bf, v_bf, o_bf);
        // out_proj + LNs
        gemm_bf16_kernel<<<dim3(Ez / 16, 1), 128, 0, stream>>>(o_bf, op_bf, out_proj_b, o2, nullptr, Ez, Ez);
        ln_kernel<<<Bz, 256, 0, stream>>>(o2, o_ln_g, o_ln_b, nullptr, u_bf, fused_bf, Ez, Dz + Ez, Dz);
        ln_kernel<<<Bz, 256, 0, stream>>>(z, z_ln_g, z_ln_b, zn, u_bf, nullptr, Dz, Dz + Ez, 0);
        // gated fusion
        gemm_bf16_kernel<<<dim3(Dz / 16, 1), 128, 0, stream>>>(u_bf, gate_bf, gate_b, gpre, nullptr, Dz, Dz + Ez);
        gemm_bf16_kernel<<<dim3(Dz / 16, 1), 128, 0, stream>>>(u_bf, cand_bf, cand_b, hpre, nullptr, Dz, Dz + Ez);
        fuse_kernel<<<(Bz * Dz) / 256, 256, 0, stream>>>(gpre, hpre, zn, fused_bf);
        // synapse MLP -> pre-activation history slot t
        gemm_bf16_kernel<<<dim3(HSz / 16, 1), 128, 0, stream>>>(fused_bf, syn1_bf, syn1_b, s1, nullptr, HSz, Dz + Ez);
        gelu_kernel<<<(Bz * HSz) / 256, 256, 0, stream>>>(s1, s1g);
        gemm_bf16_kernel<<<dim3(Dz / 16, 1), 128, 0, stream>>>(s1g, syn2_bf, syn2_b,
                                                               preh + (size_t)t * Bz * Dz, nullptr, Dz, HSz);
        // deep NLM -> new z (also writes z_ticks output)
        nlm_kernel<<<(Bz * Dz) / 256, 256, 0, stream>>>(preh, nlm1_w, nlm1_b, nlm2_w, nlm2_b,
                                                        z, out + Bz * Cz * Tz + Bz * Tz, t);
        // sync updates + head
        sync_kernel<<<(Bz * Pz) / 256, 256, 0, stream>>>(z, decays, act_left, act_right,
                                                         out_left, out_right,
                                                         a_a, b_a, a_o, b_o, s_a_bf, s_o);
        head_kernel<<<Bz, 256, 0, stream>>>(s_o, head_w, head_b, out, t);
    }
}
